// MORN_54709293416910
// MI455X (gfx1250) — matrix operand order repaired and checked by reading, not compiled, not run on hardware
//
#include <hip/hip_runtime.h>
#include <math.h>

typedef __attribute__((ext_vector_type(16))) __bf16 v16bf;
typedef __attribute__((ext_vector_type(8)))  float  v8f;

__device__ __forceinline__ float gelu_exact(float x) {
    return 0.5f * x * (1.0f + erff(x * 0.70710678118654752440f));
}

// ---------------------------------------------------------------------------
// Generic WMMA GEMM: C[M,128] = act(A[M,K] @ B[K,128] + bias)
// A,B,C f32 row-major; converted to bf16 on load for v_wmma_f32_16x16x32_bf16.
// Requires M%32==0, K%32==0, N==128. Block = 256 threads = 8 waves; wave w
// owns two 16x16 tiles at rows [blockIdx.x*32, +16) and [+16, +32), cols
// [w*16, w*16+16). Two independent WMMA accumulation chains per wave hide
// the WMMA->WMMA RAW latency and reuse the B fragment across both tiles.
// ---------------------------------------------------------------------------
__global__ __launch_bounds__(256)
void wmma_gemm_n128(const float* __restrict__ A, int K,
                    const float* __restrict__ B,
                    const float* __restrict__ bias,
                    float* __restrict__ C, int M, int act)
{
    const int N    = 128;
    int wave = threadIdx.x >> 5;
    int lane = threadIdx.x & 31;
    int m0   = blockIdx.x * 32;
    int n0   = wave * 16;
    int hs   = lane >> 4;      // lane-half select
    int lm   = lane & 15;

    const float* Arow0 = A + (size_t)(m0 + lm) * K;
    const float* Arow1 = A + (size_t)(m0 + 16 + lm) * K;
    int ncol = n0 + lm;
    v8f c0 = {}, c1 = {};
    for (int kb = 0; kb < K; kb += 32) {
        v16bf a0, a1, b;
        int ka = kb + hs * 16;
#pragma unroll
        for (int i = 0; i < 8; ++i) {
            a0[i]     = (__bf16)Arow0[ka + i];
            a0[8 + i] = (__bf16)Arow0[ka + 8 + i];
        }
#pragma unroll
        for (int i = 0; i < 8; ++i) {
            a1[i]     = (__bf16)Arow1[ka + i];
            a1[8 + i] = (__bf16)Arow1[ka + 8 + i];
        }
        int kbB = kb + hs * 16;
#pragma unroll
        for (int j = 0; j < 8; ++j) {
            b[2 * j]     = (__bf16)B[(size_t)(kbB + 2 * j) * N + ncol];
            b[2 * j + 1] = (__bf16)B[(size_t)(kbB + 2 * j + 1) * N + ncol];
        }
        if (kb + 32 < K) {   // prefetch next K-step (lowers to global_prefetch_b8)
            __builtin_prefetch(Arow0 + ka + 32, 0, 3);
            __builtin_prefetch(Arow1 + ka + 32, 0, 3);
            __builtin_prefetch(&B[(size_t)(kbB + 32) * N + ncol], 0, 3);
        }
        c0 = __builtin_amdgcn_wmma_f32_16x16x32_bf16(false, a0, false, b,
                                                     (short)0, c0, false, false);
        c1 = __builtin_amdgcn_wmma_f32_16x16x32_bf16(false, a1, false, b,
                                                     (short)0, c1, false, false);
    }
    float bv = bias ? bias[ncol] : 0.0f;
#pragma unroll
    for (int r = 0; r < 8; ++r) {
        float v0 = c0[r] + bv;
        float v1 = c1[r] + bv;
        if (act == 1) { v0 = gelu_exact(v0); v1 = gelu_exact(v1); }
        C[(size_t)(m0 + r + hs * 8) * N + ncol]      = v0;
        C[(size_t)(m0 + 16 + r + hs * 8) * N + ncol] = v1;
    }
}

// ---------------------------------------------------------------------------
// Block-diagonal per-head GEMM: Y[:, h*32:(h+1)*32] = X[:, h*32:(h+1)*32] @ W[h]
// W: 4 heads of 32x32 row-major. grid = (M/16, 8): y = head*2 + colHalf.
// One wave per block; single 16x16x32 WMMA.
// ---------------------------------------------------------------------------
__global__ __launch_bounds__(32)
void wmma_blockdiag(const float* __restrict__ X, const float* __restrict__ W,
                    float* __restrict__ Y, int M)
{
    int lane = threadIdx.x & 31;
    int head = blockIdx.y >> 1, half = blockIdx.y & 1;
    int m0 = blockIdx.x * 16;
    int lm = lane & 15, hs = lane >> 4;

    const float* Xr = X + (size_t)(m0 + lm) * 128 + head * 32;
    const float* Wh = W + head * 1024;
    v16bf a, b;
    int ka = hs * 16;
#pragma unroll
    for (int i = 0; i < 8; ++i) a[i]     = (__bf16)Xr[ka + i];
#pragma unroll
    for (int i = 0; i < 8; ++i) a[8 + i] = (__bf16)Xr[ka + 8 + i];
    int kbB = hs * 16;
    int ncol = half * 16 + lm;
#pragma unroll
    for (int j = 0; j < 8; ++j) {
        b[2 * j]     = (__bf16)Wh[(kbB + 2 * j) * 32 + ncol];
        b[2 * j + 1] = (__bf16)Wh[(kbB + 2 * j + 1) * 32 + ncol];
    }
    v8f c = {};
    c = __builtin_amdgcn_wmma_f32_16x16x32_bf16(false, a, false, b,
                                                (short)0, c, false, false);
#pragma unroll
    for (int r = 0; r < 8; ++r)
        Y[(size_t)(m0 + r + hs * 8) * 128 + head * 32 + ncol] = c[r];
}

// ---------------------------------------------------------------------------
// Small utility kernels
// ---------------------------------------------------------------------------
__global__ void fill_zero_f32(float* p, long n) {
    long i = (long)blockIdx.x * blockDim.x + threadIdx.x;
    if (i < n) p[i] = 0.0f;
}
__global__ void fill_zero_u32(unsigned* p, long n) {
    long i = (long)blockIdx.x * blockDim.x + threadIdx.x;
    if (i < n) p[i] = 0u;
}
__global__ void gather_rows(const float* __restrict__ emb, const int* __restrict__ nid,
                            float* __restrict__ h, long n128) {
    long i = (long)blockIdx.x * blockDim.x + threadIdx.x;
    if (i < n128) {
        long r = i >> 7; int d = (int)(i & 127);
        h[i] = emb[(size_t)nid[r] * 128 + d];
    }
}
__global__ void gelu_scale_k(const float* __restrict__ agg, float inv,
                             float* __restrict__ g, long n) {
    long i = (long)blockIdx.x * blockDim.x + threadIdx.x;
    if (i < n) g[i] = gelu_exact(agg[i] * inv);
}

// ---------------------------------------------------------------------------
// HGT edge passes (segment softmax with atomics; tables are L2-resident)
// ---------------------------------------------------------------------------
__device__ __forceinline__ unsigned f32_key(float f) {
    unsigned u = __float_as_uint(f);
    return (u & 0x80000000u) ? ~u : (u | 0x80000000u);
}
__global__ void edge_scores(const float* __restrict__ kk, const float* __restrict__ q,
                            const int* __restrict__ src, const int* __restrict__ dst,
                            const float* __restrict__ pri, float scale,
                            float* __restrict__ tbuf, unsigned* __restrict__ segmax, int E)
{
    int idx = blockIdx.x * blockDim.x + threadIdx.x;
    if (idx >= E * 4) return;
    int e = idx >> 2, h = idx & 3;
    int s = src[e], d = dst[e];
    const float* kr = kk + (size_t)s * 128 + h * 32;
    const float* qr = q  + (size_t)d * 128 + h * 32;
    float acc = 0.0f;
#pragma unroll
    for (int i = 0; i < 32; ++i) acc += kr[i] * qr[i];
    float t = acc * pri[h] * scale;
    tbuf[idx] = t;
    atomicMax(&segmax[(size_t)d * 4 + h], f32_key(t));
}
__global__ void edge_exp(const int* __restrict__ dst, float* __restrict__ tbuf,
                         const unsigned* __restrict__ segmax, float* __restrict__ segsum, int E)
{
    int idx = blockIdx.x * blockDim.x + threadIdx.x;
    if (idx >= E * 4) return;
    int e = idx >> 2, h = idx & 3;
    int d = dst[e];
    unsigned key = segmax[(size_t)d * 4 + h];
    float m = 0.0f;
    if (key != 0u) {
        unsigned u = (key & 0x80000000u) ? (key & 0x7fffffffu) : ~key;
        m = __uint_as_float(u);
        if (!isfinite(m)) m = 0.0f;   // reference clamps non-finite max to 0
    }
    float ev = expf(tbuf[idx] - m);
    tbuf[idx] = ev;
    atomicAdd(&segsum[(size_t)d * 4 + h], ev);
}
__global__ void edge_scatter(const float* __restrict__ vv, const int* __restrict__ src,
                             const int* __restrict__ dst, const float* __restrict__ w,
                             const float* __restrict__ tbuf, const float* __restrict__ segsum,
                             float* __restrict__ agg, int E)
{
    long idx = (long)blockIdx.x * blockDim.x + threadIdx.x;
    if (idx >= (long)E * 128) return;
    long e = idx >> 7; int dch = (int)(idx & 127); int h = dch >> 5;
    int s = src[e], d = dst[e];
    float a = tbuf[e * 4 + h] / (segsum[(size_t)d * 4 + h] + 1e-9f) * w[e];
    atomicAdd(&agg[(size_t)d * 128 + dch], vv[(size_t)s * 128 + dch] * a);
}

// ---------------------------------------------------------------------------
// Skip + LayerNorm (block per row of 128)
// ---------------------------------------------------------------------------
__global__ __launch_bounds__(128)
void skip_layernorm(const float* __restrict__ out, const float* __restrict__ h,
                    const float* __restrict__ skip_s, const float* __restrict__ gamma,
                    const float* __restrict__ beta, float* __restrict__ hn)
{
    __shared__ float red[128];
    int r = blockIdx.x, d = threadIdx.x;
    float alpha = 1.0f / (1.0f + expf(-skip_s[0]));
    float x = out[(size_t)r * 128 + d] * alpha + h[(size_t)r * 128 + d] * (1.0f - alpha);
    red[d] = x; __syncthreads();
    for (int s = 64; s > 0; s >>= 1) { if (d < s) red[d] += red[d + s]; __syncthreads(); }
    float mu = red[0] / 128.0f; __syncthreads();
    float dv = x - mu;
    red[d] = dv * dv; __syncthreads();
    for (int s = 64; s > 0; s >>= 1) { if (d < s) red[d] += red[d + s]; __syncthreads(); }
    float var = red[0] / 128.0f;
    hn[(size_t)r * 128 + d] = dv * rsqrtf(var + 1e-5f) * gamma[d] + beta[d];
}

// ---------------------------------------------------------------------------
// Column mean over M rows of a (M,128) matrix -> out[128]
// ---------------------------------------------------------------------------
__global__ __launch_bounds__(256)
void col_mean(const float* __restrict__ X, int M, float* __restrict__ out)
{
    __shared__ float red[256];
    int d = blockIdx.x;
    float acc = 0.0f;
    for (int r = threadIdx.x; r < M; r += 256) acc += X[(size_t)r * 128 + d];
    red[threadIdx.x] = acc; __syncthreads();
    for (int s = 128; s > 0; s >>= 1) {
        if (threadIdx.x < s) red[threadIdx.x] += red[threadIdx.x + s];
        __syncthreads();
    }
    if (threadIdx.x == 0) out[d] = red[0] / (float)M;
}

// ---------------------------------------------------------------------------
// WSI patch attention: block per patient, 128 threads (= K_PATCH = HID)
// ---------------------------------------------------------------------------
__global__ __launch_bounds__(128)
void wsi_attn(const float* __restrict__ qv, const float* __restrict__ kp,
              const float* __restrict__ vp, const unsigned char* __restrict__ mask,
              float* __restrict__ wtok)
{
    __shared__ float q_s[128], attn[128], red[128];
    int n = blockIdx.x, t = threadIdx.x;
    q_s[t] = qv[(size_t)n * 128 + t]; __syncthreads();
    const float* kr = kp + ((size_t)n * 128 + t) * 128;
    float acc = 0.0f;
    for (int d = 0; d < 128; ++d) acc += q_s[d] * kr[d];
    float sc = acc * 0.08838834764831845f;       // 1/sqrt(128)
    if (!mask[(size_t)n * 128 + t]) sc = -1e9f;
    red[t] = sc; __syncthreads();
    for (int s = 64; s > 0; s >>= 1) { if (t < s) red[t] = fmaxf(red[t], red[t + s]); __syncthreads(); }
    float mx = red[0]; __syncthreads();
    float ex = expf(sc - mx);
    red[t] = ex; __syncthreads();
    for (int s = 64; s > 0; s >>= 1) { if (t < s) red[t] += red[t + s]; __syncthreads(); }
    float sum = red[0]; __syncthreads();
    attn[t] = ex / sum; __syncthreads();
    const float* vb = vp + (size_t)n * 128 * 128;
    float o = 0.0f;
    for (int k = 0; k < 128; ++k) o += attn[k] * vb[(size_t)k * 128 + t];
    wtok[(size_t)n * 128 + t] = o;
}

// ---------------------------------------------------------------------------
// Token mixer + FFN + LN + logits head. Block per patient, 192 threads.
// ---------------------------------------------------------------------------
__global__ __launch_bounds__(192)
void mixer_head(const float* __restrict__ toks, const float* __restrict__ wtok,
                const float* __restrict__ Wqkv,
                const float* __restrict__ fW1, const float* __restrict__ fb1,
                const float* __restrict__ fW2, const float* __restrict__ fb2,
                const float* __restrict__ lng, const float* __restrict__ lnb,
                const float* __restrict__ gW1, const float* __restrict__ gb1,
                const float* __restrict__ gW2, const float* __restrict__ gb2,
                float* __restrict__ outL)
{
    __shared__ float tok[3][128], qkv[3][192], am[3][3];
    __shared__ float mm[3][64], h1[3][64], f2[3][64], mmn[3][64];
    __shared__ float fused[128], hid[32];
    int n = blockIdx.x, t = threadIdx.x;
    if (t < 128) {
        tok[0][t] = toks[t];                      // cnv mean
        tok[1][t] = toks[128 + t];                // rna mean
        tok[2][t] = wtok[(size_t)n * 128 + t];    // wsi token
    }
    __syncthreads();
    for (int tt = 0; tt < 3; ++tt) {
        float acc = 0.0f;
        for (int d = 0; d < 128; ++d) acc += tok[tt][d] * Wqkv[d * 192 + t];
        qkv[tt][t] = acc;
    }
    __syncthreads();
    if (t < 9) {
        int ti = t / 3, si = t % 3;
        float acc = 0.0f;
        for (int d = 0; d < 64; ++d) acc += qkv[ti][d] * qkv[si][64 + d];
        am[ti][si] = acc * 0.125f;                // 1/sqrt(64)
    }
    __syncthreads();
    if (t < 3) {
        float m = fmaxf(am[t][0], fmaxf(am[t][1], am[t][2]));
        float e0 = expf(am[t][0] - m), e1 = expf(am[t][1] - m), e2 = expf(am[t][2] - m);
        float s = e0 + e1 + e2;
        am[t][0] = e0 / s; am[t][1] = e1 / s; am[t][2] = e2 / s;
    }
    __syncthreads();
    { int tt = t >> 6, j = t & 63;
      mm[tt][j] = am[tt][0] * qkv[0][128 + j] + am[tt][1] * qkv[1][128 + j]
                + am[tt][2] * qkv[2][128 + j]; }
    __syncthreads();
    { int tt = t >> 6, j = t & 63;
      float acc = fb1[j];
      for (int i = 0; i < 64; ++i) acc += mm[tt][i] * fW1[i * 64 + j];
      h1[tt][j] = gelu_exact(acc); }
    __syncthreads();
    { int tt = t >> 6, j = t & 63;
      float acc = fb2[j];
      for (int i = 0; i < 64; ++i) acc += h1[tt][i] * fW2[i * 64 + j];
      f2[tt][j] = acc; }
    __syncthreads();
    if (t < 3) {
        float mu = 0.0f;
        for (int j = 0; j < 64; ++j) mu += f2[t][j];
        mu /= 64.0f;
        float var = 0.0f;
        for (int j = 0; j < 64; ++j) { float dv = f2[t][j] - mu; var += dv * dv; }
        var /= 64.0f;
        float inv = rsqrtf(var + 1e-5f);
        for (int j = 0; j < 64; ++j) mmn[t][j] = (f2[t][j] - mu) * inv * lng[j] + lnb[j];
    }
    __syncthreads();
    if (t < 64)       fused[t]      = 0.5f * (mmn[0][t] + mmn[1][t]);
    else if (t < 128) fused[t]      = mmn[2][t - 64];
    __syncthreads();
    if (t < 32) {
        float acc = gb1[t];
        for (int d = 0; d < 128; ++d) acc += fused[d] * gW1[d * 32 + t];
        hid[t] = fmaxf(acc, 0.0f);
    }
    __syncthreads();
    if (t < 4) {
        float acc = gb2[t];
        for (int i = 0; i < 32; ++i) acc += hid[i] * gW2[i * 4 + t];
        outL[(size_t)n * 4 + t] = acc;
    }
}

// ---------------------------------------------------------------------------
// Host driver
// ---------------------------------------------------------------------------
static inline unsigned gr(long n, int b) { return (unsigned)((n + b - 1) / b); }

extern "C" void kernel_launch(void* const* d_in, const int* in_sizes, int n_in,
                              void* d_out, int out_size, void* d_ws, size_t ws_size,
                              hipStream_t stream)
{
    (void)n_in; (void)out_size; (void)ws_size;
    const int NG = in_sizes[0];                       // 20000
    const int NP = in_sizes[14] / (128 * 1024);       // 512
    const long NG128 = (long)NG * 128;
    const long NP128 = (long)NP * 128;
    const float SCALE = 0.17677669529663687f;         // 1/sqrt(32)

    auto F = [&](int i) { return (const float*)d_in[i]; };
    auto I = [&](int i) { return (const int*)d_in[i]; };
    // param index helpers (d_in layout: 16 data inputs then params in dict order)
    auto ntP  = [&](int l, int nt, int w) { return 18 + l * 45 + nt * 11 + w; };
    auto relP = [&](int l, int r,  int w) { return 18 + l * 45 + 33 + r * 3 + w; };

    // workspace bump allocator
    float* ws = (float*)d_ws;
    size_t off = 0;
    auto alloc = [&](long n) { float* p = ws + off; off += (size_t)n; return p; };

    float* hR[2] = { alloc(NG128), alloc(NG128) };
    float* hC[2] = { alloc(NG128), alloc(NG128) };
    float* hP[2] = { alloc(NP128), alloc(NP128) };
    float* Kb[3] = { alloc(NG128), alloc(NG128), alloc(NP128) };
    float* Qb[3] = { alloc(NG128), alloc(NG128), alloc(NP128) };
    float* Vb[3] = { alloc(NG128), alloc(NG128), alloc(NP128) };
    float* kkbuf = alloc(NG128);
    float* vvbuf = alloc(NG128);
    float* Agg[3] = { alloc(NG128), alloc(NG128), alloc(NP128) };
    float* gbuf = alloc(NG128);
    float* obuf = alloc(NG128);
    long Emax = 0;
    for (int r = 0; r < 4; ++r) if (in_sizes[2 + r * 3] > Emax) Emax = in_sizes[2 + r * 3];
    float* tbuf = alloc(Emax * 4);
    unsigned* segmax = (unsigned*)alloc((long)NG * 4);
    float* segsum = alloc((long)NG * 4);
    long MPAT = (long)NP * 128;                       // 65536 patch rows
    float* pbuf  = alloc(MPAT * 128);
    float* kpbuf = alloc(MPAT * 128);
    float* vpbuf = alloc(MPAT * 128);
    float* qvbuf = alloc(NP128);
    float* wtok  = alloc(NP128);
    float* toksb = alloc(256);

    // --- initial node features ---
    gather_rows<<<gr(NG128, 256), 256, 0, stream>>>(F(16), I(0), hR[0], NG128);
    gather_rows<<<gr(NG128, 256), 256, 0, stream>>>(F(17), I(1), hC[0], NG128);
    fill_zero_f32<<<gr(NP128, 256), 256, 0, stream>>>(hP[0], NP128);

    const int Mn[3]  = { NG, NG, NP };
    const int sNT[4] = { 0, 1, 0, 1 };   // src ntype per relation rr,cc,rp,cp
    const int dNT[4] = { 0, 1, 2, 2 };   // dst ntype
    const float invn[3] = { 1.0f, 1.0f, 0.5f };

    int cur = 0;
    for (int l = 0; l < 2; ++l) {
        float* Hc[3] = { hR[cur], hC[cur], hP[cur] };
        float* Hn[3] = { hR[cur ^ 1], hC[cur ^ 1], hP[cur ^ 1] };
        for (int nt = 0; nt < 3; ++nt) {
            wmma_gemm_n128<<<Mn[nt] / 32, 256, 0, stream>>>(Hc[nt], 128, F(ntP(l, nt, 0)), F(ntP(l, nt, 1)), Kb[nt], Mn[nt], 0);
            wmma_gemm_n128<<<Mn[nt] / 32, 256, 0, stream>>>(Hc[nt], 128, F(ntP(l, nt, 2)), F(ntP(l, nt, 3)), Qb[nt], Mn[nt], 0);
            wmma_gemm_n128<<<Mn[nt] / 32, 256, 0, stream>>>(Hc[nt], 128, F(ntP(l, nt, 4)), F(ntP(l, nt, 5)), Vb[nt], Mn[nt], 0);
            fill_zero_f32<<<gr((long)Mn[nt] * 128, 256), 256, 0, stream>>>(Agg[nt], (long)Mn[nt] * 128);
        }
        for (int r = 0; r < 4; ++r) {
            int E = in_sizes[2 + r * 3];
            const int* src = I(2 + r * 3);
            const int* dst = I(3 + r * 3);
            const float* w = F(4 + r * 3);
            int snt = sNT[r], dnt = dNT[r];
            wmma_blockdiag<<<dim3(Mn[snt] / 16, 8), 32, 0, stream>>>(Kb[snt], F(relP(l, r, 0)), kkbuf, Mn[snt]);
            wmma_blockdiag<<<dim3(Mn[snt] / 16, 8), 32, 0, stream>>>(Vb[snt], F(relP(l, r, 1)), vvbuf, Mn[snt]);
            fill_zero_u32<<<gr((long)Mn[dnt] * 4, 256), 256, 0, stream>>>(segmax, (long)Mn[dnt] * 4);
            fill_zero_f32<<<gr((long)Mn[dnt] * 4, 256), 256, 0, stream>>>(segsum, (long)Mn[dnt] * 4);
            edge_scores<<<gr((long)E * 4, 256), 256, 0, stream>>>(kkbuf, Qb[dnt], src, dst, F(relP(l, r, 2)), SCALE, tbuf, segmax, E);
            edge_exp<<<gr((long)E * 4, 256), 256, 0, stream>>>(dst, tbuf, segmax, segsum, E);
            edge_scatter<<<gr((long)E * 128, 256), 256, 0, stream>>>(vvbuf, src, dst, w, tbuf, segsum, Agg[dnt], E);
        }
        for (int nt = 0; nt < 3; ++nt) {
            long n = (long)Mn[nt] * 128;
            gelu_scale_k<<<gr(n, 256), 256, 0, stream>>>(Agg[nt], invn[nt], gbuf, n);
            wmma_gemm_n128<<<Mn[nt] / 32, 256, 0, stream>>>(gbuf, 128, F(ntP(l, nt, 6)), F(ntP(l, nt, 7)), obuf, Mn[nt], 0);
            skip_layernorm<<<Mn[nt], 128, 0, stream>>>(obuf, Hc[nt], F(ntP(l, nt, 8)), F(ntP(l, nt, 9)), F(ntP(l, nt, 10)), Hn[nt]);
        }
        cur ^= 1;
    }

    // --- WSI patch attention ---
    wmma_gemm_n128<<<(unsigned)(MPAT / 32), 256, 0, stream>>>(F(14), 1024, F(108), F(109), pbuf, (int)MPAT, 1);  // p = gelu(X@Wp+bp)
    wmma_gemm_n128<<<(unsigned)(MPAT / 32), 256, 0, stream>>>(pbuf, 128, F(112), F(113), kpbuf, (int)MPAT, 0);
    wmma_gemm_n128<<<(unsigned)(MPAT / 32), 256, 0, stream>>>(pbuf, 128, F(114), F(115), vpbuf, (int)MPAT, 0);
    wmma_gemm_n128<<<NP / 32, 256, 0, stream>>>(hP[cur], 128, F(110), F(111), qvbuf, NP, 0);
    wsi_attn<<<NP, 128, 0, stream>>>(qvbuf, kpbuf, vpbuf, (const unsigned char*)d_in[15], wtok);

    // --- omics token pooling + mixer head ---
    col_mean<<<128, 256, 0, stream>>>(hC[cur], NG, toksb);        // tok0 = cnv mean
    col_mean<<<128, 256, 0, stream>>>(hR[cur], NG, toksb + 128);  // tok1 = rna mean
    mixer_head<<<NP, 192, 0, stream>>>(toksb, wtok,
                                       F(116), F(117), F(118), F(119), F(120),
                                       F(121), F(122), F(123), F(124), F(125), F(126),
                                       (float*)d_out);
}
